// MaskedTransformerBlock_13391708029712
// MI455X (gfx1250) — matrix operand order repaired and checked by reading, not compiled, not run on hardware
//
#include <hip/hip_runtime.h>
#include <hip/hip_bf16.h>
#include <math.h>

#define B_   8
#define N_   1024
#define D_   768
#define H_   12
#define HD_  64
#define MLP_ 3072

typedef _Float16 f16;
typedef _Float16 v8h  __attribute__((ext_vector_type(8)));
typedef _Float16 v16h __attribute__((ext_vector_type(16)));
typedef float    v8f  __attribute__((ext_vector_type(8)));

static __device__ __forceinline__ v16h cat8(v8h a, v8h b) {
  return __builtin_shufflevector(a, b, 0,1,2,3,4,5,6,7,8,9,10,11,12,13,14,15);
}
static __device__ __forceinline__ v16h ldg32(const f16* p0, const f16* p1) {
  return cat8(*(const v8h*)p0, *(const v8h*)p1);
}
static __device__ __forceinline__ float fast_rcp(float x) {
  return __builtin_amdgcn_rcpf(x);
}

// ---------------------------------------------------------------------------
// Weight f32 [K,Nc] -> f16 transposed [Nc,K]
// ---------------------------------------------------------------------------
__global__ void wconv_kernel(const float* __restrict__ w, f16* __restrict__ wt,
                             int K, int Nc) {
  size_t idx = (size_t)blockIdx.x * blockDim.x + threadIdx.x;
  if (idx >= (size_t)K * Nc) return;
  int k = (int)(idx / Nc), n = (int)(idx % Nc);
  wt[(size_t)n * K + k] = (f16)w[idx];
}

// ---------------------------------------------------------------------------
// Block reduction helper (blockDim == 256)
// ---------------------------------------------------------------------------
static __device__ __forceinline__ float block_reduce_sum(float v, float* red) {
  int t = threadIdx.x;
  red[t] = v; __syncthreads();
  for (int o = 128; o > 0; o >>= 1) {
    if (t < o) red[t] += red[t + o];
    __syncthreads();
  }
  float r = red[0];
  __syncthreads();
  return r;
}

// ---------------------------------------------------------------------------
// LN1: h16 = LN(x)*g + b   (one row per block, 256 threads, D=768)
// ---------------------------------------------------------------------------
__global__ __launch_bounds__(256, 1)
void ln1_kernel(const float* __restrict__ x, const float* __restrict__ gw,
                const float* __restrict__ gb, f16* __restrict__ h16) {
  __shared__ float red[256];
  const size_t row = blockIdx.x;
  const float* xr = x + row * D_;
  int t = threadIdx.x;
  float v[3];
  #pragma unroll
  for (int j = 0; j < 3; ++j) v[j] = xr[t + 256 * j];
  float mean = block_reduce_sum(v[0] + v[1] + v[2], red) * (1.f / D_);
  float d2 = 0.f;
  #pragma unroll
  for (int j = 0; j < 3; ++j) { float d = v[j] - mean; d2 += d * d; }
  float var = block_reduce_sum(d2, red) * (1.f / D_);
  float rs = rsqrtf(var + 1e-5f);
  f16* hr = h16 + row * D_;
  #pragma unroll
  for (int j = 0; j < 3; ++j) {
    int c = t + 256 * j;
    hr[c] = (f16)((v[j] - mean) * rs * gw[c] + gb[c]);
  }
}

// ---------------------------------------------------------------------------
// LN2 fused with residuals: x2 = x + attn_out + hib[b]; h16 = LN(x2)*g + b
// ---------------------------------------------------------------------------
__global__ __launch_bounds__(256, 1)
void ln2_kernel(const float* __restrict__ x, const float* __restrict__ ao,
                const float* __restrict__ hib, const float* __restrict__ gw,
                const float* __restrict__ gb, float* __restrict__ x2,
                f16* __restrict__ h16) {
  __shared__ float red[256];
  const size_t row = blockIdx.x;
  const int b = (int)(row >> 10);
  const float* xr = x + row * D_;
  const float* ar = ao + row * D_;
  const float* hr2 = hib + (size_t)b * D_;
  int t = threadIdx.x;
  float v[3];
  #pragma unroll
  for (int j = 0; j < 3; ++j) {
    int c = t + 256 * j;
    v[j] = xr[c] + ar[c] + hr2[c];
  }
  float* x2r = x2 + row * D_;
  #pragma unroll
  for (int j = 0; j < 3; ++j) x2r[t + 256 * j] = v[j];
  float mean = block_reduce_sum(v[0] + v[1] + v[2], red) * (1.f / D_);
  float d2 = 0.f;
  #pragma unroll
  for (int j = 0; j < 3; ++j) { float d = v[j] - mean; d2 += d * d; }
  float var = block_reduce_sum(d2, red) * (1.f / D_);
  float rs = rsqrtf(var + 1e-5f);
  f16* hr = h16 + row * D_;
  #pragma unroll
  for (int j = 0; j < 3; ++j) {
    int c = t + 256 * j;
    hr[c] = (f16)((v[j] - mean) * rs * gw[c] + gb[c]);
  }
}

// ---------------------------------------------------------------------------
// WMMA GEMM: C[M,Nc] = A[M,K] @ W[K,Nc] + bias, A f16 row-major,
// Wt f16 stored transposed [Nc,K].
// Each wave owns a 32x64 register tile: 2 A-frags x 4 B-frags -> 8 WMMAs per
// K-step of 32 (1.5 16B-loads per WMMA). 4 waves/block => 32x256 block tile.
// Epilogues: 0=f32 store, 1=qkv scatter to q/k/vT f16, 2=GELU->f16, 3=res+f32
// ---------------------------------------------------------------------------
enum { EPI_F32 = 0, EPI_QKV = 1, EPI_GELU = 2, EPI_RES = 3 };

template <int EPI>
__global__ __launch_bounds__(128, 1)
void gemm_wmma(const f16* __restrict__ A, const f16* __restrict__ Wt,
               const float* __restrict__ bias, int M, int K, int Nc,
               float* __restrict__ Cout, const float* __restrict__ Res,
               f16* __restrict__ P0, f16* __restrict__ P1,
               f16* __restrict__ P2) {
  const int lane = threadIdx.x & 31;
  const int wave = threadIdx.x >> 5;
  const int r = lane & 15, hi = lane >> 4;
  const int n0 = (blockIdx.x * 4 + wave) * 64;
  const int m0 = blockIdx.y * 32;
  const f16* Ar0 = A + (size_t)(m0 + r) * K;
  const f16* Ar1 = A + (size_t)(m0 + 16 + r) * K;
  const f16* Wr0 = Wt + (size_t)(n0 + r) * K;
  const f16* Wr1 = Wt + (size_t)(n0 + 16 + r) * K;
  const f16* Wr2 = Wt + (size_t)(n0 + 32 + r) * K;
  const f16* Wr3 = Wt + (size_t)(n0 + 48 + r) * K;

  v8f acc[2][4] = {};
  for (int k0 = 0; k0 < K; k0 += 32) {
    const int ka = k0 + hi * 16;
    const int kb = k0 + hi * 16;
    v16h a0 = ldg32(Ar0 + ka, Ar0 + ka + 8);
    v16h a1 = ldg32(Ar1 + ka, Ar1 + ka + 8);
    v16h b0 = ldg32(Wr0 + kb, Wr0 + kb + 8);
    v16h b1 = ldg32(Wr1 + kb, Wr1 + kb + 8);
    v16h b2 = ldg32(Wr2 + kb, Wr2 + kb + 8);
    v16h b3 = ldg32(Wr3 + kb, Wr3 + kb + 8);
    acc[0][0] = __builtin_amdgcn_wmma_f32_16x16x32_f16(false, a0, false, b0, (short)0, acc[0][0], false, false);
    acc[0][1] = __builtin_amdgcn_wmma_f32_16x16x32_f16(false, a0, false, b1, (short)0, acc[0][1], false, false);
    acc[0][2] = __builtin_amdgcn_wmma_f32_16x16x32_f16(false, a0, false, b2, (short)0, acc[0][2], false, false);
    acc[0][3] = __builtin_amdgcn_wmma_f32_16x16x32_f16(false, a0, false, b3, (short)0, acc[0][3], false, false);
    acc[1][0] = __builtin_amdgcn_wmma_f32_16x16x32_f16(false, a1, false, b0, (short)0, acc[1][0], false, false);
    acc[1][1] = __builtin_amdgcn_wmma_f32_16x16x32_f16(false, a1, false, b1, (short)0, acc[1][1], false, false);
    acc[1][2] = __builtin_amdgcn_wmma_f32_16x16x32_f16(false, a1, false, b2, (short)0, acc[1][2], false, false);
    acc[1][3] = __builtin_amdgcn_wmma_f32_16x16x32_f16(false, a1, false, b3, (short)0, acc[1][3], false, false);
  }

  float bv[4];
  #pragma unroll
  for (int ni = 0; ni < 4; ++ni) bv[ni] = bias[n0 + ni * 16 + r];

  #pragma unroll
  for (int mi = 0; mi < 2; ++mi) {
    #pragma unroll
    for (int ni = 0; ni < 4; ++ni) {
      const int c = n0 + ni * 16 + r;
      #pragma unroll
      for (int i = 0; i < 8; ++i) {
        int m = m0 + mi * 16 + i + 8 * hi;
        float v = acc[mi][ni][i] + bv[ni];
        if (EPI == EPI_F32) {
          Cout[(size_t)m * Nc + c] = v;
        } else if (EPI == EPI_RES) {
          Cout[(size_t)m * Nc + c] = Res[(size_t)m * Nc + c] + v;
        } else if (EPI == EPI_GELU) {
          float ge = 0.5f * v * (1.0f + erff(v * 0.70710678118f));
          P0[(size_t)m * Nc + c] = (f16)ge;
        } else {  // EPI_QKV: scatter to q[B,H,N,64], k[B,H,N,64], vT[B,H,64,N]
          int bb = m >> 10, n = m & 1023;
          int which = c / D_;
          int rem = c - which * D_;
          int h = rem >> 6, hd = rem & 63;
          f16 hv = (f16)v;
          if (which == 0)
            P0[(((size_t)(bb * H_ + h)) * N_ + n) * HD_ + hd] = hv;
          else if (which == 1)
            P1[(((size_t)(bb * H_ + h)) * N_ + n) * HD_ + hd] = hv;
          else
            P2[(((size_t)(bb * H_ + h)) * HD_ + hd) * N_ + n] = hv;
        }
      }
    }
  }
}

// ---------------------------------------------------------------------------
// Flash attention with sigmoid gate. 4 waves/block; wave = one 16-row q tile.
// ---------------------------------------------------------------------------
__global__ __launch_bounds__(128, 1)
void attn_flash(const f16* __restrict__ qf, const f16* __restrict__ kf,
                const f16* __restrict__ vtf, const float* __restrict__ masks,
                const float* __restrict__ lam_p, const float* __restrict__ th_p,
                const float* __restrict__ mw_p, f16* __restrict__ o16) {
  __shared__ __align__(16) f16 plds[4][16 * 32];
  const int wave = threadIdx.x >> 5;
  const int lane = threadIdx.x & 31;
  const int r = lane & 15, hi = lane >> 4;
  const int b = blockIdx.z, h = blockIdx.y;
  const int q0 = (blockIdx.x * 4 + wave) * 16;
  const int bh = b * H_ + h;
  const float lam = lam_p[bh], th = th_p[bh], mw = mw_p[bh];
  const float* mrow_p = masks + (size_t)bh * N_;
  const f16* Q = qf + (size_t)bh * N_ * HD_;
  const f16* Kp = kf + (size_t)bh * N_ * HD_;
  const f16* VT = vtf + (size_t)bh * HD_ * N_;
  const float scale = 0.125f;  // HD^-0.5

  const f16* Qr = Q + (size_t)(q0 + r) * HD_;
  v16h aq0 = ldg32(Qr + hi * 16, Qr + hi * 16 + 8);
  v16h aq1 = ldg32(Qr + 32 + hi * 16, Qr + 32 + hi * 16 + 8);

  float mq[8];
  #pragma unroll
  for (int i = 0; i < 8; ++i) mq[i] = mrow_p[q0 + i + 8 * hi];

  v8f o0 = {}, o1 = {}, o2 = {}, o3 = {};
  float mrow[8], lrow[8];
  #pragma unroll
  for (int i = 0; i < 8; ++i) { mrow[i] = -1e30f; lrow[i] = 0.f; }

  f16* pl = plds[wave];

  for (int key0 = 0; key0 < N_; key0 += 32) {
    v8f s0 = {}, s1 = {};
    {
      const f16* K0 = Kp + (size_t)(key0 + r) * HD_;
      v16h b0a = ldg32(K0 + hi * 16, K0 + hi * 16 + 8);
      v16h b0b = ldg32(K0 + 32 + hi * 16, K0 + 32 + hi * 16 + 8);
      s0 = __builtin_amdgcn_wmma_f32_16x16x32_f16(false, aq0, false, b0a,
                                                  (short)0, s0, false, false);
      s0 = __builtin_amdgcn_wmma_f32_16x16x32_f16(false, aq1, false, b0b,
                                                  (short)0, s0, false, false);
      const f16* K1 = Kp + (size_t)(key0 + 16 + r) * HD_;
      v16h b1a = ldg32(K1 + hi * 16, K1 + hi * 16 + 8);
      v16h b1b = ldg32(K1 + 32 + hi * 16, K1 + 32 + hi * 16 + 8);
      s1 = __builtin_amdgcn_wmma_f32_16x16x32_f16(false, aq0, false, b1a,
                                                  (short)0, s1, false, false);
      s1 = __builtin_amdgcn_wmma_f32_16x16x32_f16(false, aq1, false, b1b,
                                                  (short)0, s1, false, false);
    }
    const float mk0 = mrow_p[key0 + r];
    const float mk1 = mrow_p[key0 + 16 + r];
    float p0v[8], p1v[8];
    #pragma unroll
    for (int i = 0; i < 8; ++i) {
      // sigmoid via fast reciprocal (no IEEE div sequence)
      float g0 = fast_rcp(1.f + __expf(-lam * (mq[i] * mk0 - th)));
      float g1 = fast_rcp(1.f + __expf(-lam * (mq[i] * mk1 - th)));
      float l0 = s0[i] * scale * g0;
      float l1 = s1[i] * scale * g1;
      float vm = fmaxf(l0, l1);
      #pragma unroll
      for (int off = 8; off; off >>= 1) vm = fmaxf(vm, __shfl_xor(vm, off, 32));
      float mn = fmaxf(mrow[i], vm);
      float corr = __expf(mrow[i] - mn);
      float e0 = __expf(l0 - mn), e1 = __expf(l1 - mn);
      float rs = e0 + e1;
      #pragma unroll
      for (int off = 8; off; off >>= 1) rs += __shfl_xor(rs, off, 32);
      lrow[i] = lrow[i] * corr + rs;
      mrow[i] = mn;
      o0[i] *= corr; o1[i] *= corr; o2[i] *= corr; o3[i] *= corr;
      p0v[i] = e0; p1v[i] = e1;
    }
    // Re-shape P (C-frag layout) -> A-frag layout through per-wave LDS.
    #pragma unroll
    for (int i = 0; i < 8; ++i) {
      pl[(i + 8 * hi) * 32 + r]      = (f16)p0v[i];
      pl[(i + 8 * hi) * 32 + 16 + r] = (f16)p1v[i];
    }
    __asm__ volatile("s_wait_dscnt 0" ::: "memory");
    v16h ap = cat8(*(const v8h*)(pl + r * 32 + hi * 16),
                   *(const v8h*)(pl + r * 32 + hi * 16 + 8));
    const f16* V0 = VT + (size_t)(0 * 16 + r) * N_ + key0 + hi * 16;
    const f16* V1 = VT + (size_t)(1 * 16 + r) * N_ + key0 + hi * 16;
    const f16* V2 = VT + (size_t)(2 * 16 + r) * N_ + key0 + hi * 16;
    const f16* V3 = VT + (size_t)(3 * 16 + r) * N_ + key0 + hi * 16;
    v16h bv0 = cat8(*(const v8h*)V0, *(const v8h*)(V0 + 8));
    v16h bv1 = cat8(*(const v8h*)V1, *(const v8h*)(V1 + 8));
    v16h bv2 = cat8(*(const v8h*)V2, *(const v8h*)(V2 + 8));
    v16h bv3 = cat8(*(const v8h*)V3, *(const v8h*)(V3 + 8));
    o0 = __builtin_amdgcn_wmma_f32_16x16x32_f16(false, ap, false, bv0, (short)0, o0, false, false);
    o1 = __builtin_amdgcn_wmma_f32_16x16x32_f16(false, ap, false, bv1, (short)0, o1, false, false);
    o2 = __builtin_amdgcn_wmma_f32_16x16x32_f16(false, ap, false, bv2, (short)0, o2, false, false);
    o3 = __builtin_amdgcn_wmma_f32_16x16x32_f16(false, ap, false, bv3, (short)0, o3, false, false);
  }

  #pragma unroll
  for (int i = 0; i < 8; ++i) {
    int n = q0 + i + 8 * hi;
    float inv = mw * fast_rcp(lrow[i]);
    size_t base = ((size_t)(b * N_ + n)) * D_ + h * HD_;
    o16[base + 0 * 16 + r] = (f16)(o0[i] * inv);
    o16[base + 1 * 16 + r] = (f16)(o1[i] * inv);
    o16[base + 2 * 16 + r] = (f16)(o2[i] * inv);
    o16[base + 3 * 16 + r] = (f16)(o3[i] * inv);
  }
}

// ---------------------------------------------------------------------------
// g[b,d] = mean over N of attn_out[b,n,d]   (grid=B, block=768)
// ---------------------------------------------------------------------------
__global__ __launch_bounds__(768, 1)
void mean_kernel(const float* __restrict__ ao, float* __restrict__ gOut) {
  int b = blockIdx.x, d = threadIdx.x;
  float s = 0.f;
  for (int n = 0; n < N_; ++n) s += ao[((size_t)(b * N_ + n)) * D_ + d];
  gOut[(size_t)b * D_ + d] = s * (1.f / N_);
}

// ---------------------------------------------------------------------------
// Tiny head-level attention (per batch): qkv2 = g @ hi_qkv_w + b; softmax 12x12
// ---------------------------------------------------------------------------
__global__ __launch_bounds__(256, 1)
void hi_attn_kernel(const float* __restrict__ gIn, const float* __restrict__ w,
                    const float* __restrict__ bia, float* __restrict__ iv) {
  __shared__ float gS[D_];
  __shared__ float q2S[H_ * 3 * HD_];  // 12 x 192
  __shared__ float sS[H_ * H_];
  __shared__ float aS[H_ * H_];
  int b = blockIdx.x, t = threadIdx.x;
  for (int i = t; i < D_; i += 256) gS[i] = gIn[(size_t)b * D_ + i];
  __syncthreads();
  for (int i = t; i < H_ * 192; i += 256) {
    int h = i / 192, j = i % 192;
    float s = bia[j];
    for (int k = 0; k < HD_; ++k) s += gS[h * HD_ + k] * w[k * 192 + j];
    q2S[i] = s;
  }
  __syncthreads();
  for (int i = t; i < H_ * H_; i += 256) {
    int h = i / H_, g2 = i % H_;
    float s = 0.f;
    for (int d = 0; d < HD_; ++d) s += q2S[h * 192 + d] * q2S[g2 * 192 + 64 + d];
    sS[i] = s * 0.125f;
  }
  __syncthreads();
  if (t < H_) {
    float mx = -1e30f;
    for (int j = 0; j < H_; ++j) mx = fmaxf(mx, sS[t * H_ + j]);
    float sum = 0.f;
    for (int j = 0; j < H_; ++j) {
      float e = __expf(sS[t * H_ + j] - mx);
      aS[t * H_ + j] = e; sum += e;
    }
    float inv = fast_rcp(sum);
    for (int j = 0; j < H_; ++j) aS[t * H_ + j] *= inv;
  }
  __syncthreads();
  for (int i = t; i < D_; i += 256) {
    int h = i / HD_, d = i % HD_;
    float s = 0.f;
    for (int g2 = 0; g2 < H_; ++g2) s += aS[h * H_ + g2] * q2S[g2 * 192 + 128 + d];
    iv[(size_t)b * D_ + i] = s;
  }
}

// ---------------------------------------------------------------------------
// hib[b,j] = iv_flat[b] @ hi_proj_w + hi_proj_b   (rank-1 per batch, tiny)
// ---------------------------------------------------------------------------
__global__ void hib_kernel(const float* __restrict__ iv,
                           const float* __restrict__ w,
                           const float* __restrict__ bia,
                           float* __restrict__ hib) {
  int idx = blockIdx.x * blockDim.x + threadIdx.x;
  if (idx >= B_ * D_) return;
  int b = idx / D_, j = idx % D_;
  float s = bia[j];
  const float* ivb = iv + (size_t)b * D_;
  for (int k = 0; k < D_; ++k) s += ivb[k] * w[(size_t)k * D_ + j];
  hib[idx] = s;
}

// ---------------------------------------------------------------------------
__global__ void tail_kernel(const float* __restrict__ mw, float* __restrict__ out) {
  int t = threadIdx.x;
  if (t < B_ * H_) out[(size_t)B_ * N_ * D_ + t] = mw[t];
}

// ---------------------------------------------------------------------------
extern "C" void kernel_launch(void* const* d_in, const int* in_sizes, int n_in,
                              void* d_out, int out_size, void* d_ws,
                              size_t ws_size, hipStream_t stream) {
  const float* x      = (const float*)d_in[0];
  const float* masks  = (const float*)d_in[1];
  const float* lam    = (const float*)d_in[2];
  const float* theta  = (const float*)d_in[3];
  const float* mwts   = (const float*)d_in[4];
  const float* ln1_g  = (const float*)d_in[5];
  const float* ln1_b  = (const float*)d_in[6];
  const float* qkv_w  = (const float*)d_in[7];
  const float* qkv_b  = (const float*)d_in[8];
  const float* proj_w = (const float*)d_in[9];
  const float* proj_b = (const float*)d_in[10];
  const float* hi_qkv_w  = (const float*)d_in[11];
  const float* hi_qkv_b  = (const float*)d_in[12];
  const float* hi_proj_w = (const float*)d_in[13];
  const float* hi_proj_b = (const float*)d_in[14];
  const float* ln2_g  = (const float*)d_in[15];
  const float* ln2_b  = (const float*)d_in[16];
  const float* fc1_w  = (const float*)d_in[17];
  const float* fc1_b  = (const float*)d_in[18];
  const float* fc2_w  = (const float*)d_in[19];
  const float* fc2_b  = (const float*)d_in[20];
  float* out = (float*)d_out;

  const size_t M = (size_t)B_ * N_;  // 8192 rows

  char* ws = (char*)d_ws;
  size_t off = 0;
  auto alloc = [&](size_t bytes) -> char* {
    off = (off + 255) & ~(size_t)255;
    char* p = ws + off;
    off += bytes;
    return p;
  };

  f16* qkvT = (f16*)alloc((size_t)3 * D_ * D_ * 2);        // [2304,768]
  f16* projT = (f16*)alloc((size_t)D_ * D_ * 2);           // [768,768]
  f16* fc1T = (f16*)alloc((size_t)MLP_ * D_ * 2);          // [3072,768]
  f16* fc2T = (f16*)alloc((size_t)D_ * MLP_ * 2);          // [768,3072]
  f16* hbuf = (f16*)alloc(M * D_ * 2);                     // h16 / h2_16
  char* big = alloc(4 * M * D_ * 2);                       // q,k,vT,o16 / m16
  f16* qf  = (f16*)big;
  f16* kf  = (f16*)(big + 1 * M * D_ * 2);
  f16* vtf = (f16*)(big + 2 * M * D_ * 2);
  f16* o16 = (f16*)(big + 3 * M * D_ * 2);
  f16* m16 = (f16*)big;                                    // reuse after attn
  float* attn_out = (float*)alloc(M * D_ * 4);
  float* x2 = (float*)alloc(M * D_ * 4);
  float* gbuf = (float*)alloc((size_t)B_ * D_ * 4);
  float* ivb  = (float*)alloc((size_t)B_ * D_ * 4);
  float* hibb = (float*)alloc((size_t)B_ * D_ * 4);
  (void)ws_size; (void)in_sizes; (void)n_in; (void)out_size;

  // 1) Weight transpose-converts to f16
  {
    size_t t;
    t = (size_t)D_ * 3 * D_;
    wconv_kernel<<<(t + 255) / 256, 256, 0, stream>>>(qkv_w, qkvT, D_, 3 * D_);
    t = (size_t)D_ * D_;
    wconv_kernel<<<(t + 255) / 256, 256, 0, stream>>>(proj_w, projT, D_, D_);
    t = (size_t)D_ * MLP_;
    wconv_kernel<<<(t + 255) / 256, 256, 0, stream>>>(fc1_w, fc1T, D_, MLP_);
    t = (size_t)MLP_ * D_;
    wconv_kernel<<<(t + 255) / 256, 256, 0, stream>>>(fc2_w, fc2T, MLP_, D_);
  }

  // 2) LN1
  ln1_kernel<<<(int)M, 256, 0, stream>>>(x, ln1_g, ln1_b, hbuf);

  // 3) qkv GEMM with scatter epilogue  (block tile 32x256)
  gemm_wmma<EPI_QKV><<<dim3((3 * D_) / 256, (int)(M / 32)), 128, 0, stream>>>(
      hbuf, qkvT, qkv_b, (int)M, D_, 3 * D_, nullptr, nullptr, qf, kf, vtf);

  // 4) gated flash attention
  attn_flash<<<dim3(N_ / 64, H_, B_), 128, 0, stream>>>(
      qf, kf, vtf, masks, lam, theta, mwts, o16);

  // 5) output projection
  gemm_wmma<EPI_F32><<<dim3(D_ / 256, (int)(M / 32)), 128, 0, stream>>>(
      o16, projT, proj_b, (int)M, D_, D_, attn_out, nullptr, nullptr, nullptr,
      nullptr);

  // 6) head-mean, tiny head attention, hib
  mean_kernel<<<B_, D_, 0, stream>>>(attn_out, gbuf);
  hi_attn_kernel<<<B_, 256, 0, stream>>>(gbuf, hi_qkv_w, hi_qkv_b, ivb);
  hib_kernel<<<(B_ * D_ + 255) / 256, 256, 0, stream>>>(ivb, hi_proj_w,
                                                        hi_proj_b, hibb);

  // 7) residuals + LN2
  ln2_kernel<<<(int)M, 256, 0, stream>>>(x, attn_out, hibb, ln2_g, ln2_b, x2,
                                         hbuf);

  // 8) fc1 + GELU (f16 out into reused big buffer)
  gemm_wmma<EPI_GELU><<<dim3(MLP_ / 256, (int)(M / 32)), 128, 0, stream>>>(
      hbuf, fc1T, fc1_b, (int)M, D_, MLP_, nullptr, nullptr, m16, nullptr,
      nullptr);

  // 9) fc2 + residual -> d_out
  gemm_wmma<EPI_RES><<<dim3(D_ / 256, (int)(M / 32)), 128, 0, stream>>>(
      m16, fc2T, fc2_b, (int)M, MLP_, D_, out, x2, nullptr, nullptr, nullptr);

  // 10) mask_weights passthrough
  tail_kernel<<<1, 128, 0, stream>>>(mwts, out);
}
